// RecurrentModel_69793218560590
// MI455X (gfx1250) — compile-verified, matrix operand order checked by reading, not run on hardware
//
#include <hip/hip_runtime.h>
#include <hip/hip_bf16.h>

typedef __attribute__((ext_vector_type(16))) __bf16 bf16x16;
typedef __attribute__((ext_vector_type(8)))  __bf16 bf16x8;
typedef __attribute__((ext_vector_type(8)))  float  f32x8;

#define WMMA_BF16(a, b, c) \
  __builtin_amdgcn_wmma_f32_16x16x32_bf16(false, (a), false, (b), (short)0, (c), false, false)

static constexpr int Bb   = 32;
static constexpr int Tt   = 1024;
static constexpr int DIN  = 512;
static constexpr int AA   = 256;
static constexpr int HH   = 512;
static constexpr int VV   = 41;
static constexpr int MR   = Bb * Tt;   // 32768 rows (b,t)
static constexpr int G3H  = 3 * HH;    // 1536

// ---------------------------------------------------------------------------
// CDNA5 async global->LDS copy (ASYNCcnt-tracked, 16B per lane per op).
// VGPR operand 0 = LDS byte offset, operand 1 = 64-bit global address.
// ---------------------------------------------------------------------------
__device__ __forceinline__ void async_copy_b128(unsigned lds_off, unsigned long long gaddr) {
  asm volatile("global_load_async_to_lds_b128 %0, %1, off"
               :: "v"(lds_off), "v"(gaddr) : "memory");
}
__device__ __forceinline__ void wait_async0() {
  asm volatile("s_wait_asynccnt 0" ::: "memory");
}

// Fast saturating gate math: v_exp_f32 + v_rcp_f32, no IEEE div chain.
__device__ __forceinline__ float fast_sigmoid(float x) {
  return __builtin_amdgcn_rcpf(1.0f + __builtin_amdgcn_exp2f(-1.4426950408889634f * x));
}
__device__ __forceinline__ float fast_tanh(float x) {
  const float e = __builtin_amdgcn_exp2f(2.8853900817779268f * x);
  return 1.0f - 2.0f * __builtin_amdgcn_rcpf(e + 1.0f);
}

// ---------------------------------------------------------------------------
// Fragment loader for V_WMMA_F32_16X16X32_BF16 operands from a row-major
// matrix [*, ld] (bf16). Lane L covers row (L%16); VGPRs 0-3 hold
// K = kb..kb+7, VGPRs 4-7 hold K = kb+16..kb+23, kb = 8*(L/16).
// ---------------------------------------------------------------------------
__device__ __forceinline__ bf16x16 frag_load(const __bf16* __restrict__ base,
                                             int row, int ld, int k0, int lane) {
  const int kb = (lane >> 4) << 3;
  const __bf16* p = base + (size_t)row * ld + k0 + kb;
  bf16x8 lo = *(const bf16x8*)p;
  bf16x8 hi = *(const bf16x8*)(p + 16);
  bf16x16 f;
#pragma unroll
  for (int i = 0; i < 8; ++i) { f[i] = lo[i]; f[8 + i] = hi[i]; }
  return f;
}

// ---------------------------------------------------------------------------
__global__ void k_f32_to_bf16(const float* __restrict__ src,
                              __bf16* __restrict__ dst, int n) {
  for (int i = blockIdx.x * blockDim.x + threadIdx.x; i < n;
       i += gridDim.x * blockDim.x)
    dst[i] = (__bf16)src[i];
}

// W_fc (41x512) -> bf16 padded to 48x512 (pad rows = 0)
__global__ void k_prep_wfc(const float* __restrict__ src, __bf16* __restrict__ dst) {
  int i = blockIdx.x * blockDim.x + threadIdx.x;
  if (i < 48 * 512) {
    int v = i >> 9;
    dst[i] = (v < VV) ? (__bf16)src[i] : (__bf16)0.0f;
  }
}

__global__ void k_init(__bf16* __restrict__ hbuf, unsigned* __restrict__ cnt) {
  int i = blockIdx.x * blockDim.x + threadIdx.x;
  if (i < Bb * HH) hbuf[i] = (__bf16)0.0f;
  if (i == 0) *cnt = 0u;
}

// ---------------------------------------------------------------------------
// Generic WMMA GEMM: OUT[M,N] = X[M,K](bf16) @ W[N,K]^T(bf16) + bias[N]
// One 16x16 tile per wave, dual fp32 accumulators to halve the D->C chain.
// ---------------------------------------------------------------------------
template <int KDIM, typename OT>
__global__ __launch_bounds__(256) void k_gemm(const __bf16* __restrict__ X,
                                              const __bf16* __restrict__ W,
                                              const float* __restrict__ bias,
                                              OT* __restrict__ out,
                                              int Mtiles, int Ntiles) {
  const int wave = threadIdx.x >> 5;
  const int lane = threadIdx.x & 31;
  const int tile = blockIdx.x * 8 + wave;
  if (tile >= Mtiles * Ntiles) return;           // wave-uniform, EXEC stays full
  const int nt = tile % Ntiles;
  const int mt = tile / Ntiles;
  const int arow = (mt << 4) + (lane & 15);
  const int bcol = (nt << 4) + (lane & 15);

  f32x8 acc0, acc1;
#pragma unroll
  for (int r = 0; r < 8; ++r) { acc0[r] = 0.0f; acc1[r] = 0.0f; }

  for (int k0 = 0; k0 < KDIM; k0 += 64) {
    acc0 = WMMA_BF16(frag_load(X, arow, KDIM, k0, lane),
                     frag_load(W, bcol, KDIM, k0, lane), acc0);
    acc1 = WMMA_BF16(frag_load(X, arow, KDIM, k0 + 32, lane),
                     frag_load(W, bcol, KDIM, k0 + 32, lane), acc1);
  }

  const int rbase = (lane >> 4) << 3;            // D: row = vgpr + 8*(lane/16)
  const int N = Ntiles << 4;
  const float bv = bias[bcol];
#pragma unroll
  for (int r = 0; r < 8; ++r)
    out[(size_t)((mt << 4) + rbase + r) * N + bcol] = (OT)(acc0[r] + acc1[r] + bv);
}

// ---------------------------------------------------------------------------
// GRU recurrence. 16 workgroups, WG w owns hidden units j in [32w, 32w+32),
// i.e. W_hh rows {j, 512+j, 1024+j} -> 96x512 bf16 staged ONCE in LDS via
// async global->LDS b128 copies. Per step: 12 waves x one 16x16 WMMA tile,
// gate math on own 32 units, h broadcast via L2 ping-pong + 16-WG barrier.
// ---------------------------------------------------------------------------
__global__ __launch_bounds__(384) void k_gru(const float* __restrict__ gx,
                                             const __bf16* __restrict__ Whh,
                                             const float* __restrict__ bhh,
                                             __bf16* __restrict__ hbuf,   // [2][32*512]
                                             __bf16* __restrict__ hs,     // [32768*512]
                                             unsigned* __restrict__ cnt) {
  __shared__ __attribute__((aligned(16))) __bf16 sW[96 * 512];  // 98304 B
  __shared__ float sGH[32 * 96];                                // 12 KB
  __shared__ float sH[32 * 32];                                 // own h slice, fp32
  __shared__ float sBhh[96];

  const int w    = blockIdx.x;          // 0..15
  const int tid  = threadIdx.x;         // 0..383
  const int lane = tid & 31;
  const int wave = tid >> 5;            // 0..11

  // --- one-time W_hh slice staging: async global->LDS (ASYNCcnt path) ---
  const unsigned sw_base = (unsigned)(size_t)(&sW[0]);
  for (int c = tid; c < 96 * 64; c += 384) {            // 16B chunks
    const int lr = c >> 6, kc = (c & 63) << 4;          // byte offset in row
    const int R  = (lr >> 5) * 512 + (w << 5) + (lr & 31);
    async_copy_b128(sw_base + (unsigned)(lr * 1024 + kc),
                    (unsigned long long)(size_t)Whh + (unsigned long long)R * 1024 + kc);
  }
  for (int i = tid; i < 96; i += 384) {
    const int g = i >> 5, jl = i & 31;
    sBhh[i] = bhh[g * 512 + (w << 5) + jl];
  }
  for (int i = tid; i < 32 * 32; i += 384) sH[i] = 0.0f;
  wait_async0();
  __syncthreads();

  const int mt    = wave & 1;           // batch tile (0..1)
  const int nt    = wave >> 1;          // gate-col tile (0..5)
  const int arow  = (mt << 4) + (lane & 15);
  const int bcol  = (nt << 4) + (lane & 15);
  const int rbase = (lane >> 4) << 3;

  for (int t = 0; t < Tt; ++t) {
    // prefetch this step's gate pre-activations so they overlap the WMMAs
#pragma unroll 1
    for (int i = tid; i < 32 * 32; i += 384) {
      const float* gxp = gx + ((size_t)(i >> 5) * Tt + t) * G3H + (w << 5) + (i & 31);
      __builtin_prefetch(gxp, 0, 0);
      __builtin_prefetch(gxp + HH, 0, 0);
      __builtin_prefetch(gxp + 2 * HH, 0, 0);
    }

    const __bf16* hcur = hbuf + (size_t)(t & 1) * (Bb * HH);
    f32x8 acc0, acc1;
#pragma unroll
    for (int r = 0; r < 8; ++r) { acc0[r] = 0.0f; acc1[r] = 0.0f; }
#pragma unroll 2
    for (int k0 = 0; k0 < HH; k0 += 64) {
      acc0 = WMMA_BF16(frag_load(hcur, arow, HH, k0, lane),
                       frag_load(sW,   bcol, HH, k0, lane), acc0);
      acc1 = WMMA_BF16(frag_load(hcur, arow, HH, k0 + 32, lane),
                       frag_load(sW,   bcol, HH, k0 + 32, lane), acc1);
    }
#pragma unroll
    for (int r = 0; r < 8; ++r)
      sGH[((mt << 4) + rbase + r) * 96 + bcol] = acc0[r] + acc1[r];
    __syncthreads();

    __bf16* hnext = hbuf + (size_t)((t + 1) & 1) * (Bb * HH);
    for (int i = tid; i < 32 * 32; i += 384) {
      const int bi = i >> 5, jl = i & 31;
      const int j  = (w << 5) + jl;
      const float* gxp = gx + ((size_t)bi * Tt + t) * G3H;
      const float ghr = sGH[bi * 96 + jl]      + sBhh[jl];
      const float ghz = sGH[bi * 96 + 32 + jl] + sBhh[32 + jl];
      const float ghn = sGH[bi * 96 + 64 + jl] + sBhh[64 + jl];
      const float rg = fast_sigmoid(gxp[j] + ghr);
      const float zg = fast_sigmoid(gxp[HH + j] + ghz);
      const float ng = fast_tanh(gxp[2 * HH + j] + rg * ghn);
      const float hnew = (1.0f - zg) * ng + zg * sH[i];
      sH[i] = hnew;
      hs[((size_t)bi * Tt + t) * HH + j] = (__bf16)hnew;
      hnext[bi * HH + j] = (__bf16)hnew;
    }

    // device-wide barrier across the 16 resident workgroups
    __syncthreads();
    if (tid == 0) {
      __threadfence();
      atomicAdd(cnt, 1u);
      const unsigned target = 16u * (unsigned)(t + 1);
      while (__hip_atomic_load(cnt, __ATOMIC_ACQUIRE, __HIP_MEMORY_SCOPE_AGENT) < target) {}
      __threadfence();
    }
    __syncthreads();
  }
}

// ---------------------------------------------------------------------------
// logits = hs @ W_fc^T + b_fc (N padded 41->48), then log_softmax over 41.
// ---------------------------------------------------------------------------
__global__ __launch_bounds__(256) void k_logits(const __bf16* __restrict__ hs,
                                                const __bf16* __restrict__ wfc,
                                                const float* __restrict__ bfc,
                                                float* __restrict__ out) {
  __shared__ float sL[8 * 16 * 48];     // 24 KB
  const int tid = threadIdx.x, lane = tid & 31, wave = tid >> 5;
  const int mt    = blockIdx.x * 8 + wave;
  const int arow  = (mt << 4) + (lane & 15);
  const int ncol  = lane & 15;
  const int rbase = (lane >> 4) << 3;

  f32x8 acc[3];
#pragma unroll
  for (int n = 0; n < 3; ++n) {
    const int colg = (n << 4) + ncol;
    const float bv = (colg < VV) ? bfc[colg] : 0.0f;
#pragma unroll
    for (int r = 0; r < 8; ++r) acc[n][r] = bv;
  }

  for (int k0 = 0; k0 < HH; k0 += 32) {
    bf16x16 a = frag_load(hs, arow, HH, k0, lane);
#pragma unroll
    for (int n = 0; n < 3; ++n) {
      bf16x16 b = frag_load(wfc, (n << 4) + ncol, HH, k0, lane);
      acc[n] = WMMA_BF16(a, b, acc[n]);
    }
  }

  float* sLw = sL + wave * 768;
#pragma unroll
  for (int n = 0; n < 3; ++n)
#pragma unroll
    for (int r = 0; r < 8; ++r)
      sLw[(rbase + r) * 48 + (n << 4) + ncol] = acc[n][r];
  __syncthreads();

  for (int rr = tid; rr < 128; rr += blockDim.x) {
    const float* Lrow = sL + (rr >> 4) * 768 + (rr & 15) * 48;
    float mx = -1e30f;
    for (int v = 0; v < VV; ++v) mx = fmaxf(mx, Lrow[v]);
    float s = 0.0f;
    for (int v = 0; v < VV; ++v) s += __builtin_amdgcn_exp2f(1.4426950408889634f * (Lrow[v] - mx));
    const float lse = mx + 0.6931471805599453f * __builtin_amdgcn_logf(s);
    const size_t m = (size_t)blockIdx.x * 128 + rr;
    for (int v = 0; v < VV; ++v) out[m * VV + v] = Lrow[v] - lse;
  }
}

// ---------------------------------------------------------------------------
extern "C" void kernel_launch(void* const* d_in, const int* in_sizes, int n_in,
                              void* d_out, int out_size, void* d_ws, size_t ws_size,
                              hipStream_t stream) {
  const float* x   = (const float*)d_in[0];   // (32,1024,512)
  const float* Wa  = (const float*)d_in[1];   // (256,512)
  const float* ba  = (const float*)d_in[2];   // (256,)
  const float* Wih = (const float*)d_in[3];   // (1536,256)
  const float* Whh = (const float*)d_in[4];   // (1536,512)
  const float* bih = (const float*)d_in[5];   // (1536,)
  const float* bhh = (const float*)d_in[6];   // (1536,)
  const float* Wfc = (const float*)d_in[7];   // (41,512)
  const float* bfc = (const float*)d_in[8];   // (41,)
  float* out = (float*)d_out;                 // (32,1024,41) fp32

  char* ws = (char*)d_ws;
  size_t off = 0;
  auto take = [&](size_t bytes) -> void* {
    void* p = (void*)(ws + off);
    off += (bytes + 255) & ~(size_t)255;
    return p;
  };

  __bf16* xb   = (__bf16*)take(sizeof(__bf16) * (size_t)MR * DIN);
  __bf16* Wab  = (__bf16*)take(sizeof(__bf16) * (size_t)AA * DIN);
  __bf16* Wihb = (__bf16*)take(sizeof(__bf16) * (size_t)G3H * AA);
  __bf16* Whhb = (__bf16*)take(sizeof(__bf16) * (size_t)G3H * HH);
  __bf16* Wfcb = (__bf16*)take(sizeof(__bf16) * (size_t)48 * HH);
  __bf16* ab   = (__bf16*)take(sizeof(__bf16) * (size_t)MR * AA);
  float*  gxf  = (float*) take(sizeof(float)  * (size_t)MR * G3H);
  __bf16* hbuf = (__bf16*)take(sizeof(__bf16) * (size_t)2 * Bb * HH);
  __bf16* hsb  = (__bf16*)take(sizeof(__bf16) * (size_t)MR * HH);
  unsigned* cnt = (unsigned*)take(64);

  // ---- precision conversion (fp32 -> bf16) ----
  k_f32_to_bf16<<<2048, 256, 0, stream>>>(x,   xb,   MR * DIN);
  k_f32_to_bf16<<<512,  256, 0, stream>>>(Wa,  Wab,  AA * DIN);
  k_f32_to_bf16<<<1024, 256, 0, stream>>>(Wih, Wihb, G3H * AA);
  k_f32_to_bf16<<<2048, 256, 0, stream>>>(Whh, Whhb, G3H * HH);
  k_prep_wfc   <<<96,   256, 0, stream>>>(Wfc, Wfcb);
  k_init       <<<64,   256, 0, stream>>>(hbuf, cnt);

  // ---- adapter: a = x @ Wa^T + ba  (32768x512 @ 512x256) ----
  k_gemm<DIN, __bf16><<<(MR / 16) * (AA / 16) / 8, 256, 0, stream>>>(
      xb, Wab, ba, ab, MR / 16, AA / 16);

  // ---- gate pre-activations: gx = a @ Wih^T + bih (32768x256 @ 256x1536) ----
  k_gemm<AA, float><<<(MR / 16) * (G3H / 16) / 8, 256, 0, stream>>>(
      ab, Wihb, bih, gxf, MR / 16, G3H / 16);

  // ---- sequential GRU scan: 16 cooperating workgroups ----
  k_gru<<<16, 384, 0, stream>>>(gxf, Whhb, bhh, hbuf, hsb, cnt);

  // ---- FC + log_softmax ----
  k_logits<<<MR / 128, 256, 0, stream>>>(hsb, Wfcb, bfc, out);
}
